// RadixMLPQwen3Attention_39075612459263
// MI455X (gfx1250) — compile-verified, matrix operand order checked by reading, not run on hardware
//
#include <hip/hip_runtime.h>
#include <cstddef>

// ---------------------------------------------------------------------------
// Problem constants (match reference)
// ---------------------------------------------------------------------------
#define NC_  6144
#define T_   8192
#define B_   8
#define S_   1024
#define H_   16
#define KV_  8
#define D_   128
#define HID_ 2048

// ---------------------------------------------------------------------------
// WMMA vector types (CDNA5 gfx1250, wave32)
// ---------------------------------------------------------------------------
typedef __attribute__((ext_vector_type(16))) __bf16 v16bf;
typedef __attribute__((ext_vector_type(8)))  float  v8f;

struct alignas(16) U16x8 { unsigned short s[8]; };

union Frag16 {
    v16bf          v;
    U16x8          h[2];
    unsigned short s[16];
};

// fp32 -> bf16 (round to nearest even), self-contained bit math
__device__ __forceinline__ unsigned short f2bf(float f) {
    union { float f; unsigned int u; } v; v.f = f;
    unsigned int r = v.u + 0x7FFFu + ((v.u >> 16) & 1u);
    return (unsigned short)(r >> 16);
}

// Load a 16(row) x 32(K) bf16 fragment in the CDNA5 A/B register layout:
//   lanes 0-15 : row = lane,    K = {0..7, 16..23}
//   lanes 16-31: row = lane-16, K = {8..15, 24..31}
// `base` points at element [0][0] of the tile, `stride` = row stride in elems.
__device__ __forceinline__ v16bf load_frag_g(const unsigned short* base, int stride) {
    const int lane = threadIdx.x & 31;
    const int r    = lane & 15;
    const unsigned short* p = base + r * stride + ((lane >> 4) << 3);
    Frag16 f;
    f.h[0] = *(const U16x8*)(p);        // K = k0 .. k0+7
    f.h[1] = *(const U16x8*)(p + 16);   // K = k0+16 .. k0+23
    return f.v;
}

// ---------------------------------------------------------------------------
// fp32 -> bf16 elementwise conversion
// ---------------------------------------------------------------------------
__global__ void k_f2bf(const float* __restrict__ in, unsigned short* __restrict__ out, int n) {
    int i = blockIdx.x * 256 + threadIdx.x;
    if (i < n) out[i] = f2bf(in[i]);
}

// ---------------------------------------------------------------------------
// Generic GEMM: C[M,N] = X[M,K] @ W[N,K]^T, bf16 in, fp32 out.
// One wave (32 threads) per block computes a 32x64 tile of C
// (2 M-frags x 4 N-frags, 8 accumulators -> 8 WMMAs per 6 frag loads).
// grid = (N/64, M/32)
// ---------------------------------------------------------------------------
__global__ void k_gemm_bf16(const unsigned short* __restrict__ X,
                            const unsigned short* __restrict__ W,
                            float* __restrict__ C, int M, int N, int K) {
    const int m0   = blockIdx.y << 5;
    const int n0   = blockIdx.x << 6;
    const int lane = threadIdx.x & 31;

    v8f acc[2][4];
#pragma unroll
    for (int mi = 0; mi < 2; ++mi)
#pragma unroll
        for (int j = 0; j < 4; ++j) acc[mi][j] = (v8f){};

    const unsigned short* x0 = X + (size_t)m0 * K;
    const unsigned short* x1 = X + (size_t)(m0 + 16) * K;
    const unsigned short* w0 = W + (size_t)(n0     ) * K;
    const unsigned short* w1 = W + (size_t)(n0 + 16) * K;
    const unsigned short* w2 = W + (size_t)(n0 + 32) * K;
    const unsigned short* w3 = W + (size_t)(n0 + 48) * K;

    for (int k = 0; k < K; k += 32) {
        __builtin_prefetch(x0 + k + 128, 0, 1);   // global_prefetch_b8
        __builtin_prefetch(w0 + k + 128, 0, 1);
        v16bf a0 = load_frag_g(x0 + k, K);
        v16bf a1 = load_frag_g(x1 + k, K);
        v16bf b0 = load_frag_g(w0 + k, K);
        v16bf b1 = load_frag_g(w1 + k, K);
        v16bf b2 = load_frag_g(w2 + k, K);
        v16bf b3 = load_frag_g(w3 + k, K);
        acc[0][0] = __builtin_amdgcn_wmma_f32_16x16x32_bf16(false, a0, false, b0, (short)0, acc[0][0], false, false);
        acc[0][1] = __builtin_amdgcn_wmma_f32_16x16x32_bf16(false, a0, false, b1, (short)0, acc[0][1], false, false);
        acc[0][2] = __builtin_amdgcn_wmma_f32_16x16x32_bf16(false, a0, false, b2, (short)0, acc[0][2], false, false);
        acc[0][3] = __builtin_amdgcn_wmma_f32_16x16x32_bf16(false, a0, false, b3, (short)0, acc[0][3], false, false);
        acc[1][0] = __builtin_amdgcn_wmma_f32_16x16x32_bf16(false, a1, false, b0, (short)0, acc[1][0], false, false);
        acc[1][1] = __builtin_amdgcn_wmma_f32_16x16x32_bf16(false, a1, false, b1, (short)0, acc[1][1], false, false);
        acc[1][2] = __builtin_amdgcn_wmma_f32_16x16x32_bf16(false, a1, false, b2, (short)0, acc[1][2], false, false);
        acc[1][3] = __builtin_amdgcn_wmma_f32_16x16x32_bf16(false, a1, false, b3, (short)0, acc[1][3], false, false);
    }

    // C layout: VGPR i, lanes 0-15 -> (M=i, N=lane); lanes 16-31 -> (M=i+8, N=lane-16)
    const int col = n0 + (lane & 15);
    const int hi  = (lane >> 4) << 3;
#pragma unroll
    for (int mi = 0; mi < 2; ++mi) {
#pragma unroll
        for (int i = 0; i < 8; ++i) {
            float* c = C + (size_t)(m0 + mi * 16 + hi + i) * N + col;
            c[0]  = acc[mi][0][i];
            c[16] = acc[mi][1][i];
            c[32] = acc[mi][2][i];
            c[48] = acc[mi][3][i];
        }
    }
}

// ---------------------------------------------------------------------------
// Fused RMSNorm + RoPE, in place over x[tok][head][128].
// grid = (ntok, nheads), block = 32 (one wave per head vector)
// ---------------------------------------------------------------------------
__global__ void k_rmsnorm_rope(float* __restrict__ x,
                               const float* __restrict__ cs,
                               const float* __restrict__ sn,
                               const float* __restrict__ nw, int nheads) {
    const int tok  = blockIdx.x;
    const int head = blockIdx.y;
    const int lane = threadIdx.x;
    float* p = x + ((size_t)tok * nheads + head) * D_;

    float x0 = p[lane], x1 = p[lane + 32], x2 = p[lane + 64], x3 = p[lane + 96];
    float ss = x0 * x0 + x1 * x1 + x2 * x2 + x3 * x3;
#pragma unroll
    for (int off = 1; off < 32; off <<= 1) ss += __shfl_xor(ss, off, 32);
    const float inv = rsqrtf(ss * (1.0f / 128.0f) + 1e-6f);

    x0 *= inv * nw[lane];
    x1 *= inv * nw[lane + 32];
    x2 *= inv * nw[lane + 64];
    x3 *= inv * nw[lane + 96];

    const float c0 = cs[(size_t)tok * 64 + lane],      s0 = sn[(size_t)tok * 64 + lane];
    const float c1 = cs[(size_t)tok * 64 + lane + 32], s1 = sn[(size_t)tok * 64 + lane + 32];

    p[lane]      = x0 * c0 - x2 * s0;
    p[lane + 64] = x0 * s0 + x2 * c0;
    p[lane + 32] = x1 * c1 - x3 * s1;
    p[lane + 96] = x1 * s1 + x3 * c1;
}

// ---------------------------------------------------------------------------
// Scatter (gather by scatter_indices) + bf16 convert; Q pre-scaled by D^-1/2.
// grid = T_, block = 256
// ---------------------------------------------------------------------------
__global__ void k_scatter_bf16(const float* __restrict__ qf,
                               const float* __restrict__ kf,
                               const float* __restrict__ vf,
                               const int* __restrict__ idx,
                               unsigned short* __restrict__ qs,
                               unsigned short* __restrict__ ks,
                               unsigned short* __restrict__ vs, float scale) {
    const int t   = blockIdx.x;
    const int src = idx[t];
    const int tid = threadIdx.x;
#pragma unroll
    for (int e = tid; e < H_ * D_; e += 256)
        qs[(size_t)t * (H_ * D_) + e] = f2bf(qf[(size_t)src * (H_ * D_) + e] * scale);
#pragma unroll
    for (int e = tid; e < KV_ * D_; e += 256) {
        ks[(size_t)t * (KV_ * D_) + e] = f2bf(kf[(size_t)src * (KV_ * D_) + e]);
        vs[(size_t)t * (KV_ * D_) + e] = f2bf(vf[(size_t)src * (KV_ * D_) + e]);
    }
}

// ---------------------------------------------------------------------------
// Causal GQA flash attention. One wave per (b, h, 16-row q tile).
// KV loop steps by 32 rows: two 16x16 score tiles per step, 32-wide online
// softmax, and a FULL-K (16x32 @ 32x128) PV WMMA (no zero padding).
// grid = (S/16, H, B), block = 32. Sequences are full (len == S).
// ---------------------------------------------------------------------------
__global__ void k_flash_attn(const unsigned short* __restrict__ qs,
                             const unsigned short* __restrict__ ks,
                             const unsigned short* __restrict__ vs,
                             float* __restrict__ out) {
    __shared__ U16x8          sV[32 * 16];   // 32 x 128 bf16 V tile (8 KB)
    __shared__ unsigned short sP[16 * 32];   // 16 x 32 bf16 P tile (1 KB)

    const int qt   = blockIdx.x;
    const int h    = blockIdx.y;
    const int b    = blockIdx.z;
    const int lane = threadIdx.x;
    const int kvh  = h >> 1;                 // H/KV == 2
    const int Ncol = lane & 15;
    const int Mhi  = (lane >> 4) << 3;
    const int k0   = (lane < 16) ? 0 : 8;
    const int r    = lane & 15;
    const int qrow0 = qt * 16;

    // Q fragments (pre-scaled bf16), row stride = H*D
    const unsigned short* Qb = qs + ((size_t)(b * S_ + qrow0) * H_ + h) * D_;
    v16bf aq[4];
#pragma unroll
    for (int d = 0; d < 4; ++d) aq[d] = load_frag_g(Qb + d * 32, H_ * D_);

    v8f   acc[8];
    float mrow[8], lrow[8];
#pragma unroll
    for (int n = 0; n < 8; ++n) acc[n] = (v8f){};
#pragma unroll
    for (int i = 0; i < 8; ++i) { mrow[i] = -3.0e38f; lrow[i] = 0.0f; }

    const int nb = (qt >> 1) + 1;            // ceil((qt+1)*16 / 32)
    for (int jb = 0; jb < nb; ++jb) {
        const int kv0 = jb * 32;
        const unsigned short* Kb0 = ks + ((size_t)(b * S_ + kv0) * KV_ + kvh) * D_;
        const unsigned short* Kb1 = Kb0 + (size_t)16 * KV_ * D_;
        const unsigned short* Vb0 = vs + ((size_t)(b * S_ + kv0) * KV_ + kvh) * D_;

        // Two 16x16 score tiles: S = Q K^T
        v8f sc0 = (v8f){}, sc1 = (v8f){};
#pragma unroll
        for (int d = 0; d < 4; ++d) {
            v16bf bk0 = load_frag_g(Kb0 + d * 32, KV_ * D_);
            sc0 = __builtin_amdgcn_wmma_f32_16x16x32_bf16(false, aq[d], false, bk0, (short)0, sc0, false, false);
        }
#pragma unroll
        for (int d = 0; d < 4; ++d) {
            v16bf bk1 = load_frag_g(Kb1 + d * 32, KV_ * D_);
            sc1 = __builtin_amdgcn_wmma_f32_16x16x32_bf16(false, aq[d], false, bk1, (short)0, sc1, false, false);
        }

        // Stage 32x128 V tile into LDS (512 x 16B chunks, 16 per lane)
        __syncthreads();
#pragma unroll
        for (int c = 0; c < 16; ++c) {
            const int chunk = lane + c * 32;
            const int row   = chunk >> 4;
            const int col   = chunk & 15;
            sV[chunk] = *(const U16x8*)(Vb0 + (size_t)row * (KV_ * D_) + col * 8);
        }

        // Causal mask (wave-uniform predicates; only last tiles ever need it)
        if (kv0 + 15 > qrow0) {
#pragma unroll
            for (int i = 0; i < 8; ++i)
                if (kv0 + Ncol > qrow0 + Mhi + i) sc0[i] = -3.0e38f;
        }
        if (kv0 + 31 > qrow0) {
#pragma unroll
            for (int i = 0; i < 8; ++i)
                if (kv0 + 16 + Ncol > qrow0 + Mhi + i) sc1[i] = -3.0e38f;
        }

        // Row max across 32 columns (pairwise, then half-wave shuffle)
        float rmax[8];
#pragma unroll
        for (int i = 0; i < 8; ++i) rmax[i] = fmaxf(sc0[i], sc1[i]);
#pragma unroll
        for (int off = 1; off < 16; off <<= 1) {
#pragma unroll
            for (int i = 0; i < 8; ++i)
                rmax[i] = fmaxf(rmax[i], __shfl_xor(rmax[i], off, 32));
        }

        float p0[8], p1[8], alpha[8], rsum[8];
#pragma unroll
        for (int i = 0; i < 8; ++i) {
            const float mn = fmaxf(mrow[i], rmax[i]);
            alpha[i] = __expf(mrow[i] - mn);
            mrow[i]  = mn;
            p0[i]    = __expf(sc0[i] - mn);
            p1[i]    = __expf(sc1[i] - mn);
            rsum[i]  = p0[i] + p1[i];
        }
#pragma unroll
        for (int off = 1; off < 16; off <<= 1) {
#pragma unroll
            for (int i = 0; i < 8; ++i) rsum[i] += __shfl_xor(rsum[i], off, 32);
        }
#pragma unroll
        for (int i = 0; i < 8; ++i) lrow[i] = lrow[i] * alpha[i] + rsum[i];
#pragma unroll
        for (int n = 0; n < 8; ++n)
#pragma unroll
            for (int i = 0; i < 8; ++i) acc[n][i] *= alpha[i];

        // P tile (16x32) -> LDS (re-layout C-format to A-format)
#pragma unroll
        for (int i = 0; i < 8; ++i) {
            sP[(Mhi + i) * 32 + Ncol]      = f2bf(p0[i]);
            sP[(Mhi + i) * 32 + 16 + Ncol] = f2bf(p1[i]);
        }
        __syncthreads();

        // Full-K A fragment of P (K = 0..31 all live)
        Frag16 fp;
#pragma unroll
        for (int i = 0; i < 8; ++i) {
            fp.s[i]     = sP[r * 32 + k0 + i];
            fp.s[8 + i] = sP[r * 32 + k0 + 16 + i];
        }

        // O += P @ V over 8 column slices of D (full 32-deep K)
        const unsigned short* sVs = (const unsigned short*)sV;
#pragma unroll
        for (int n = 0; n < 8; ++n) {
            Frag16 fv;
#pragma unroll
            for (int i = 0; i < 8; ++i) {
                fv.s[i]     = sVs[(k0 + i)      * 128 + n * 16 + Ncol];
                fv.s[8 + i] = sVs[(k0 + 16 + i) * 128 + n * 16 + Ncol];
            }
            acc[n] = __builtin_amdgcn_wmma_f32_16x16x32_bf16(false, fp.v, false, fv.v, (short)0, acc[n], false, false);
        }
    }

    // Normalize and store: out[t][h][d], fp32
    float invl[8];
#pragma unroll
    for (int i = 0; i < 8; ++i) invl[i] = 1.0f / lrow[i];
    float* Ob = out + ((size_t)(b * S_ + qrow0) * H_ + h) * D_;
#pragma unroll
    for (int n = 0; n < 8; ++n)
#pragma unroll
        for (int i = 0; i < 8; ++i)
            Ob[(size_t)(Mhi + i) * (H_ * D_) + n * 16 + Ncol] = acc[n][i] * invl[i];
}

// ---------------------------------------------------------------------------
// fold_gather + bf16 convert: ac[i] = bf16(attn[fold_gather[i]])
// grid = NC_, block = 256
// ---------------------------------------------------------------------------
__global__ void k_fold_bf16(const float* __restrict__ attn,
                            const int* __restrict__ fold,
                            unsigned short* __restrict__ ac) {
    const int i   = blockIdx.x;
    const int src = fold[i];
    const int tid = threadIdx.x;
#pragma unroll
    for (int e = tid; e < HID_; e += 256)
        ac[(size_t)i * HID_ + e] = f2bf(attn[(size_t)src * HID_ + e]);
}

// ---------------------------------------------------------------------------
// Host-side launch
// ---------------------------------------------------------------------------
static inline size_t align256(size_t x) { return (x + 255) & ~(size_t)255; }

extern "C" void kernel_launch(void* const* d_in, const int* in_sizes, int n_in,
                              void* d_out, int out_size, void* d_ws, size_t ws_size,
                              hipStream_t stream) {
    (void)in_sizes; (void)n_in; (void)out_size; (void)ws_size;

    const float* hidden = (const float*)d_in[0];   // NC x HID
    const float* cosb   = (const float*)d_in[1];   // NC x 64
    const float* sinb   = (const float*)d_in[2];   // NC x 64
    // d_in[3] = cu_seq_lengths (uniform, unused), d_in[4] = max_seq_len (unused)
    const int*   fold    = (const int*)d_in[5];    // NC
    const int*   scatter = (const int*)d_in[6];    // T
    const float* q_w = (const float*)d_in[7];      // 2048 x 2048
    const float* k_w = (const float*)d_in[8];      // 1024 x 2048
    const float* v_w = (const float*)d_in[9];      // 1024 x 2048
    const float* o_w = (const float*)d_in[10];     // 2048 x 2048
    const float* qnw = (const float*)d_in[11];     // 128
    const float* knw = (const float*)d_in[12];     // 128

    // Workspace carve-up (all 256B aligned)
    char* w = (char*)d_ws;
    size_t off = 0;
    auto take = [&](size_t bytes) { char* p = w + off; off += align256(bytes); return p; };

    unsigned short* xbf  = (unsigned short*)take((size_t)NC_ * HID_ * 2);
    unsigned short* qwbf = (unsigned short*)take((size_t)H_  * D_ * HID_ * 2);
    unsigned short* kwbf = (unsigned short*)take((size_t)KV_ * D_ * HID_ * 2);
    unsigned short* vwbf = (unsigned short*)take((size_t)KV_ * D_ * HID_ * 2);
    unsigned short* owbf = (unsigned short*)take((size_t)HID_ * H_ * D_ * 2);
    float*          qf   = (float*)take((size_t)NC_ * H_  * D_ * 4);
    float*          kf   = (float*)take((size_t)NC_ * KV_ * D_ * 4);
    float*          vf   = (float*)take((size_t)NC_ * KV_ * D_ * 4);
    unsigned short* qsb  = (unsigned short*)take((size_t)T_ * H_  * D_ * 2);
    unsigned short* ksb  = (unsigned short*)take((size_t)T_ * KV_ * D_ * 2);
    unsigned short* vsb  = (unsigned short*)take((size_t)T_ * KV_ * D_ * 2);
    float*          attn = (float*)take((size_t)T_ * H_ * D_ * 4);
    unsigned short* acbf = (unsigned short*)take((size_t)NC_ * HID_ * 2);

    // 1) fp32 -> bf16 conversions
    auto conv = [&](const float* src, unsigned short* dst, int n) {
        k_f2bf<<<(n + 255) / 256, 256, 0, stream>>>(src, dst, n);
    };
    conv(hidden, xbf,  NC_ * HID_);
    conv(q_w,    qwbf, H_  * D_ * HID_);
    conv(k_w,    kwbf, KV_ * D_ * HID_);
    conv(v_w,    vwbf, KV_ * D_ * HID_);
    conv(o_w,    owbf, HID_ * H_ * D_);

    // 2) Q/K/V projections (C[M,N] = X @ W^T), 32x64 tile per wave
    k_gemm_bf16<<<dim3((H_  * D_) / 64, NC_ / 32), 32, 0, stream>>>(xbf, qwbf, qf, NC_, H_  * D_, HID_);
    k_gemm_bf16<<<dim3((KV_ * D_) / 64, NC_ / 32), 32, 0, stream>>>(xbf, kwbf, kf, NC_, KV_ * D_, HID_);
    k_gemm_bf16<<<dim3((KV_ * D_) / 64, NC_ / 32), 32, 0, stream>>>(xbf, vwbf, vf, NC_, KV_ * D_, HID_);

    // 3) RMSNorm + RoPE (in place)
    k_rmsnorm_rope<<<dim3(NC_, H_ ), 32, 0, stream>>>(qf, cosb, sinb, qnw, H_);
    k_rmsnorm_rope<<<dim3(NC_, KV_), 32, 0, stream>>>(kf, cosb, sinb, knw, KV_);

    // 4) Scatter into (B,S) order + bf16 (Q pre-scaled by 1/sqrt(D))
    const float scale = 0.08838834764831845f; // 128^-0.5
    k_scatter_bf16<<<T_, 256, 0, stream>>>(qf, kf, vf, scatter, qsb, ksb, vsb, scale);

    // 5) Causal GQA flash attention (32-wide KV steps, full-K PV)
    k_flash_attn<<<dim3(S_ / 16, H_, B_), 32, 0, stream>>>(qsb, ksb, vsb, attn);

    // 6) fold_gather + bf16
    k_fold_bf16<<<NC_, 256, 0, stream>>>(attn, fold, acbf);

    // 7) Output projection straight into d_out
    k_gemm_bf16<<<dim3(HID_ / 64, NC_ / 32), 32, 0, stream>>>(acbf, owbf, (float*)d_out, NC_, HID_, HID_);
}
